// FunctionalComponentAnalysis_9328668967473
// MI455X (gfx1250) — matrix operand order repaired and checked by reading, not compiled, not run on hardware
//
#include <hip/hip_runtime.h>

// ---------------------------------------------------------------------------
// Problem: out[N,R] = x[N,D] @ GramSchmidt(params[R,D]).T
//   N = 65536, D = 1024, R = 256
// Strategy (MI455X / gfx1250):
//   GS(P) == L^{-1} P with L = chol(P P^T)   (Cholesky-QR, exact-arithmetic
//   identical to sequential row-wise Gram-Schmidt, well conditioned here).
//   Main GEMM in bf16-split (hi+lo) via v_wmma_f32_16x16x32_bf16, 3 products
//   per tile for ~fp32 accuracy; double-buffered LDS with async global->LDS
//   (ASYNCcnt) so HBM latency hides behind the WMMAs. Memory floor:
//   320 MB / 23.3 TB/s ~= 14 us.
// ---------------------------------------------------------------------------

typedef __attribute__((ext_vector_type(16))) __bf16 v16bf;
typedef __attribute__((ext_vector_type(8)))  __bf16 bf16x8;
typedef __attribute__((ext_vector_type(4)))  __bf16 bf16x4;
typedef __attribute__((ext_vector_type(8)))  float  v8f;
typedef __attribute__((ext_vector_type(2)))  int    v2i_t;

#define TOK_N 65536
#define DIM_D 1024
#define RANK_R 256

// ---- CDNA5 async global->LDS (ASYNCcnt) with safe fallback -----------------
#if defined(__has_builtin)
#if __has_builtin(__builtin_amdgcn_global_load_async_to_lds_b64) && \
    __has_builtin(__builtin_amdgcn_s_wait_asynccnt)
#define HAVE_ASYNC_LDS 1
#endif
#endif

#define GAS __attribute__((address_space(1)))
#define LAS __attribute__((address_space(3)))

__device__ __forceinline__ void async_copy_b64(const void* g, void* l) {
#ifdef HAVE_ASYNC_LDS
    // arg0: v2i addrspace(1)* (clang prints AS1 as "__device__"),
    // arg1: LDS-side pointer (flat LDS addr low 32 bits == LDS byte offset).
    __builtin_amdgcn_global_load_async_to_lds_b64(
        (GAS v2i_t*)(unsigned long long)g,
        (LAS v2i_t*)(unsigned int)(unsigned long long)l,
        0, 0);
#else
    *(uint2*)l = *(const uint2*)g;
#endif
}

__device__ __forceinline__ void async_drain() {
#ifdef HAVE_ASYNC_LDS
    __builtin_amdgcn_s_wait_asynccnt(0);
#endif
}

// ---------------------------------------------------------------------------
// Kernel 1: G = P @ P^T   ([256,1024] -> [256,256])
// ---------------------------------------------------------------------------
__global__ __launch_bounds__(256)
void fca_gram(const float* __restrict__ P, float* __restrict__ G) {
    const int i = blockIdx.y * 16 + threadIdx.y;
    const int j = blockIdx.x * 16 + threadIdx.x;
    const float4* __restrict__ a = (const float4*)(P + (size_t)i * DIM_D);
    const float4* __restrict__ b = (const float4*)(P + (size_t)j * DIM_D);
    float s = 0.f;
#pragma unroll 4
    for (int k = 0; k < DIM_D / 4; ++k) {
        float4 va = a[k], vb = b[k];
        s += va.x * vb.x + va.y * vb.y + va.z * vb.z + va.w * vb.w;
    }
    G[i * RANK_R + j] = s;
}

// ---------------------------------------------------------------------------
// Kernel 2: in-place Cholesky of G (256x256) in 256KB dynamic LDS
// (fits CDNA5's 320KB WGP LDS). One workgroup, 1024 threads.
// ---------------------------------------------------------------------------
__global__ __launch_bounds__(1024)
void fca_chol(float* __restrict__ G) {
    extern __shared__ float sG[];          // 65536 floats = 256 KB
    const int tid = threadIdx.x;
    for (int idx = tid; idx < RANK_R * RANK_R / 4; idx += 1024)
        ((float4*)sG)[idx] = ((const float4*)G)[idx];
    __syncthreads();

    const int j  = tid & 255;
    const int q4 = tid >> 8;               // 0..3
    for (int k = 0; k < RANK_R; ++k) {
        if (tid == 0) sG[k * RANK_R + k] = sqrtf(sG[k * RANK_R + k]);
        __syncthreads();
        const float inv = 1.0f / sG[k * RANK_R + k];
        for (int i = k + 1 + tid; i < RANK_R; i += 1024)
            sG[i * RANK_R + k] *= inv;
        __syncthreads();
        if (j > k) {
            const float gjk = sG[j * RANK_R + k];
            for (int i = k + 1 + q4; i < RANK_R; i += 4)
                sG[i * RANK_R + j] -= sG[i * RANK_R + k] * gjk;
        }
        __syncthreads();
    }
    for (int idx = tid; idx < RANK_R * RANK_R; idx += 1024) {
        const int r = idx >> 8, c = idx & 255;
        G[idx] = (c <= r) ? sG[idx] : 0.0f;
    }
}

// ---------------------------------------------------------------------------
// Kernel 3: Q = L^{-1} P, emitted as bf16 hi/lo pair (row-major [256,1024]).
// Wave32 forward substitution, lane l owns q_j (j%32==l) in registers,
// shfl_xor tree reductions; L resident in 256KB dynamic LDS.
// ---------------------------------------------------------------------------
__global__ __launch_bounds__(256)
void fca_trsm(const float* __restrict__ P, const float* __restrict__ L,
              __bf16* __restrict__ qhi, __bf16* __restrict__ qlo) {
    extern __shared__ float sL[];          // 65536 floats = 256 KB
    const int tid = threadIdx.x;
    const int lane = tid & 31, wave = tid >> 5;
    for (int idx = tid; idx < RANK_R * RANK_R / 4; idx += 256)
        ((float4*)sL)[idx] = ((const float4*)L)[idx];
    __syncthreads();

    const int gw = blockIdx.x * 8 + wave;  // 0..255
    for (int t = 0; t < 4; ++t) {
        const int c = gw + 256 * t;        // column of P/Q, 0..1023
        float q[8];
#pragma unroll
        for (int blk = 0; blk < 8; ++blk) {
            q[blk] = 0.0f;
            for (int s = 0; s < 32; ++s) {
                const int i = blk * 32 + s;
                const float* lrow = sL + i * RANK_R + lane;
                float part = 0.0f;
#pragma unroll
                for (int b = 0; b < 8; ++b)
                    if (b < blk) part += lrow[32 * b] * q[b];
                if (lane < s) part += lrow[32 * blk] * q[blk];
#pragma unroll
                for (int m = 16; m >= 1; m >>= 1)
                    part += __shfl_xor(part, m, 32);
                const float pi = P[(size_t)i * DIM_D + c];
                const float qi = (pi - part) / sL[i * (RANK_R + 1)];
                if (lane == s) q[blk] = qi;
            }
        }
#pragma unroll
        for (int blk = 0; blk < 8; ++blk) {
            const float v = q[blk];
            const __bf16 h = (__bf16)v;
            const __bf16 l = (__bf16)(v - (float)h);
            const size_t o = (size_t)(blk * 32 + lane) * DIM_D + c;
            qhi[o] = h;
            qlo[o] = l;
        }
    }
}

// ---------------------------------------------------------------------------
// Kernel 4: out[N,256] = x[N,1024] @ Q^T, bf16-split WMMA, double-buffered LDS.
//   512 threads = 16 wave32s (4x4 grid); block tile 128x128; K staged 64 wide.
//   Per iter: issue async Q loads (next stage) -> pull next x into regs ->
//   24 WMMAs on current stage -> convert/store x(next) -> wait asynccnt ->
//   barrier. LDS: 2 stages x 4 arrays x [128][72] bf16 = 144 KB.
// ---------------------------------------------------------------------------
#define BM 128
#define BN 128
#define KB 64
#define LSTR 72                 // padded LDS row stride (bf16 elems)
#define STAGE_ELEMS (4 * BM * LSTR)

static __device__ __forceinline__ v16bf frag16(const __bf16* p, int gap) {
    bf16x8 a = *(const bf16x8*)p;
    bf16x8 b = *(const bf16x8*)(p + gap);
    return __builtin_shufflevector(a, b, 0, 1, 2, 3, 4, 5, 6, 7,
                                   8, 9, 10, 11, 12, 13, 14, 15);
}

__global__ __launch_bounds__(512)
void fca_gemm(const float* __restrict__ x, const __bf16* __restrict__ qhi,
              const __bf16* __restrict__ qlo, float* __restrict__ out) {
    extern __shared__ __bf16 smem[];       // 2 stages

    const int tid  = threadIdx.x;
    const int lane = tid & 31;
    const int wave = tid >> 5;
    const int wm   = wave >> 2;            // 0..3  (M group of 32)
    const int wn   = wave & 3;             // 0..3  (N group of 32)
    const int m0   = blockIdx.x * BM;
    const int n0   = blockIdx.y * BN;

    const int lr = tid >> 4;               // 0..31 loader row
    const int lc = (tid & 15) * 4;         // loader col (x4)

    v8f acc[2][2] = {};
    float4 xr[4];

    // stage layout: [sxh | sxl | sqh | sql], each [BM][LSTR]

    // prologue: fill stage 0
    {
        __bf16* sxh = smem;
        __bf16* sxl = sxh + BM * LSTR;
        __bf16* sqh = sxl + BM * LSTR;
        __bf16* sql = sqh + BN * LSTR;
#pragma unroll
        for (int p = 0; p < 4; ++p) {
            const int r = p * 32 + lr;
            const size_t qo = (size_t)(n0 + r) * DIM_D + lc;
            async_copy_b64(qhi + qo, &sqh[r * LSTR + lc]);
            async_copy_b64(qlo + qo, &sql[r * LSTR + lc]);
            const float4 v = *(const float4*)(x + (size_t)(m0 + r) * DIM_D + lc);
            const __bf16 h0 = (__bf16)v.x, h1 = (__bf16)v.y;
            const __bf16 h2 = (__bf16)v.z, h3 = (__bf16)v.w;
            bf16x4 hv = {h0, h1, h2, h3};
            bf16x4 lv = {(__bf16)(v.x - (float)h0), (__bf16)(v.y - (float)h1),
                         (__bf16)(v.z - (float)h2), (__bf16)(v.w - (float)h3)};
            *(bf16x4*)(&sxh[r * LSTR + lc]) = hv;
            *(bf16x4*)(&sxl[r * LSTR + lc]) = lv;
        }
        async_drain();
        __syncthreads();
    }

    for (int kk = 0; kk < DIM_D; kk += KB) {
        const int cur = (kk / KB) & 1;
        const int nxt = cur ^ 1;
        __bf16* csxh = smem + cur * STAGE_ELEMS;
        __bf16* csxl = csxh + BM * LSTR;
        __bf16* csqh = csxl + BM * LSTR;
        __bf16* csql = csqh + BN * LSTR;
        __bf16* nsxh = smem + nxt * STAGE_ELEMS;
        __bf16* nsxl = nsxh + BM * LSTR;
        __bf16* nsqh = nsxl + BM * LSTR;
        __bf16* nsql = nsqh + BN * LSTR;

        const bool more = (kk + KB) < DIM_D;
        if (more) {
            // issue next-stage Q tiles asynchronously (no VGPR round-trip)
#pragma unroll
            for (int p = 0; p < 4; ++p) {
                const int r = p * 32 + lr;
                const size_t qo = (size_t)(n0 + r) * DIM_D + kk + KB + lc;
                async_copy_b64(qhi + qo, &nsqh[r * LSTR + lc]);
                async_copy_b64(qlo + qo, &nsql[r * LSTR + lc]);
            }
            // pull next-stage x into registers (latency hides behind WMMAs)
#pragma unroll
            for (int p = 0; p < 4; ++p)
                xr[p] = *(const float4*)(x + (size_t)(m0 + p * 32 + lr) * DIM_D +
                                         kk + KB + lc);
        }

        // ---- 24 WMMAs on current stage (two 32-deep k-subtiles) ----
#pragma unroll
        for (int ks = 0; ks < KB; ks += 32) {
            const int ak = ks + ((lane >> 4) * 8);    // A frag K offset
            const int bk = ks + ((lane >> 4) * 8);    // B frag K offset
            const int arow = wm * 32 + (lane & 15);
            const int brow = wn * 32 + (lane & 15);
            v16bf ah[2], al[2], bh[2], bl[2];
#pragma unroll
            for (int t = 0; t < 2; ++t) {
                ah[t] = frag16(&csxh[(arow + t * 16) * LSTR + ak], 16);
                al[t] = frag16(&csxl[(arow + t * 16) * LSTR + ak], 16);
                bh[t] = frag16(&csqh[(brow + t * 16) * LSTR + bk], 16);
                bl[t] = frag16(&csql[(brow + t * 16) * LSTR + bk], 16);
            }
#pragma unroll
            for (int mt = 0; mt < 2; ++mt)
#pragma unroll
                for (int nt = 0; nt < 2; ++nt) {
                    acc[mt][nt] = __builtin_amdgcn_wmma_f32_16x16x32_bf16(
                        false, ah[mt], false, bh[nt], (short)0, acc[mt][nt],
                        false, false);
                    acc[mt][nt] = __builtin_amdgcn_wmma_f32_16x16x32_bf16(
                        false, ah[mt], false, bl[nt], (short)0, acc[mt][nt],
                        false, false);
                    acc[mt][nt] = __builtin_amdgcn_wmma_f32_16x16x32_bf16(
                        false, al[mt], false, bh[nt], (short)0, acc[mt][nt],
                        false, false);
                }
        }

        if (more) {
            // convert + store next-stage x (fp32 -> bf16 hi/lo)
#pragma unroll
            for (int p = 0; p < 4; ++p) {
                const int r = p * 32 + lr;
                const float4 v = xr[p];
                const __bf16 h0 = (__bf16)v.x, h1 = (__bf16)v.y;
                const __bf16 h2 = (__bf16)v.z, h3 = (__bf16)v.w;
                bf16x4 hv = {h0, h1, h2, h3};
                bf16x4 lv = {(__bf16)(v.x - (float)h0),
                             (__bf16)(v.y - (float)h1),
                             (__bf16)(v.z - (float)h2),
                             (__bf16)(v.w - (float)h3)};
                *(bf16x4*)(&nsxh[r * LSTR + lc]) = hv;
                *(bf16x4*)(&nsxl[r * LSTR + lc]) = lv;
            }
            async_drain();
        }
        __syncthreads();
    }

    // ---- epilogue: C/D layout -> global (row = v + 8*(lane>=16)) ----
#pragma unroll
    for (int mt = 0; mt < 2; ++mt)
#pragma unroll
        for (int nt = 0; nt < 2; ++nt) {
            const int gr0 = m0 + wm * 32 + mt * 16 + ((lane >> 4) * 8);
            const int gc  = n0 + wn * 32 + nt * 16 + (lane & 15);
#pragma unroll
            for (int v = 0; v < 8; ++v)
                out[(size_t)(gr0 + v) * RANK_R + gc] = acc[mt][nt][v];
        }
}

// ---------------------------------------------------------------------------
extern "C" void kernel_launch(void* const* d_in, const int* in_sizes, int n_in,
                              void* d_out, int out_size, void* d_ws,
                              size_t ws_size, hipStream_t stream) {
    const float* x      = (const float*)d_in[0];   // [65536,1024] fp32
    const float* params = (const float*)d_in[1];   // [256,1024]  fp32
    float* out = (float*)d_out;                    // [65536,256] fp32

    char* ws = (char*)d_ws;
    float*  G   = (float*)ws;                          // 256*256 f32 (G -> L)
    __bf16* qhi = (__bf16*)(ws + 262144);              // 256*1024 bf16
    __bf16* qlo = (__bf16*)(ws + 262144 + 524288);     // 256*1024 bf16

    fca_gram<<<dim3(16, 16), dim3(16, 16), 0, stream>>>(params, G);
    fca_chol<<<1, 1024, RANK_R * RANK_R * sizeof(float), stream>>>(G);
    fca_trsm<<<32, 256, RANK_R * RANK_R * sizeof(float), stream>>>(params, G,
                                                                   qhi, qlo);
    fca_gemm<<<dim3(TOK_N / BM, RANK_R / BN), 512,
               2 * STAGE_ELEMS * sizeof(__bf16), stream>>>(x, qhi, qlo, out);
}
